// Graph_model_498216206613
// MI455X (gfx1250) — matrix operand order repaired and checked by reading, not compiled, not run on hardware
//
#include <hip/hip_runtime.h>
#include <hip/hip_bf16.h>
#include <stdint.h>

// ---------------- model constants ----------------
#define N_NODES   20000
#define N_EDGES   320000
#define NFEAT     256
#define NUM_HEAD  8
#define HEAD_D    32
#define NEG_SLOPE 0.01f
#define LN_EPS    1e-5f

// ---------------- types ----------------
typedef __bf16 bf16_t;
typedef bf16_t v16bf __attribute__((ext_vector_type(16)));
typedef float  v8f   __attribute__((ext_vector_type(8)));
typedef float  v4f   __attribute__((ext_vector_type(4)));
typedef unsigned int v4u __attribute__((ext_vector_type(4)));

union FragU { v4u u[2]; v16bf b; };

__device__ __forceinline__ float leakyf(float x) { return x >= 0.f ? x : NEG_SLOPE * x; }

__device__ __forceinline__ unsigned short f2bf(float f) {
  unsigned int u = __float_as_uint(f);
  u += 0x7FFFu + ((u >> 16) & 1u);           // round-to-nearest-even
  return (unsigned short)(u >> 16);
}
// order-preserving float <-> uint encoding for atomicMax-based segment max
__device__ __forceinline__ unsigned int encf(float f) {
  unsigned int u = __float_as_uint(f);
  return (u & 0x80000000u) ? ~u : (u | 0x80000000u);
}
__device__ __forceinline__ float decf(unsigned int x) {
  unsigned int u = (x & 0x80000000u) ? (x & 0x7FFFFFFFu) : ~x;
  return __uint_as_float(u);
}

// ---------------- weight prep: fp32 (K,N) -> bf16 transposed (N,K) ----------------
__global__ void k_prep_w(const float* __restrict__ W, unsigned short* __restrict__ Wt,
                         int K, int N) {
  int idx = blockIdx.x * blockDim.x + threadIdx.x;
  if (idx >= K * N) return;
  int k = idx / N, n = idx % N;
  Wt[(size_t)n * K + k] = f2bf(W[idx]);
}

// ---------------- node embedding gather ----------------
__global__ void k_embed(const int* __restrict__ at, const float* __restrict__ emb,
                        float* __restrict__ dst, int ldd, int n) {
  int idx = blockIdx.x * blockDim.x + threadIdx.x;
  if (idx >= n * NFEAT) return;
  int node = idx >> 8, c = idx & 255;
  dst[(size_t)node * ldd + c] = emb[(size_t)at[node] * NFEAT + c];
}

// ---------------- WMMA bf16 GEMM: C(MxN,f32) = A(MxK,f32) x Bt(NxK,bf16)^T ----------------
// 256 threads = 8 waves as 4(M) x 2(N); block tile 128x128, wave tile 32x64, K-step 32.
// A staged via VGPRs (fp32->bf16 convert); B staged with GLOBAL_LOAD_ASYNC_TO_LDS_B128.
// flags: bit0 = add bias[N], bit1 = leaky epilogue
#define BM 128
#define BN 128
#define BK 32
#define LDSROW 40   // ushorts per row (80B: 16B aligned, bank-conflict-dodging pad)

__global__ __launch_bounds__(256) void k_gemm(
    const float* __restrict__ A, int lda,
    const unsigned short* __restrict__ Bt,
    const float* __restrict__ bias,
    float* __restrict__ C, int ldc,
    int M, int N, int K, int flags)
{
  __shared__ __align__(16) unsigned short Asl[BM * LDSROW];
  __shared__ __align__(16) unsigned short Bsl[BN * LDSROW];

  const int tid  = threadIdx.x;
  const int wave = tid >> 5;
  const int lane = tid & 31;
  const int mw   = wave & 3;       // wave row group (32 rows)
  const int nw   = wave >> 2;      // wave col group (64 cols)
  const int m0 = blockIdx.y * BM;
  const int n0 = blockIdx.x * BN;

  v8f acc[2][4] = {};

  // A staging: 2 threads/row, 16 fp32 each (converted to bf16)
  const int arow = tid >> 1;
  const int aseg = (tid & 1) * 16;

  const int half = lane >> 4;
  const int fm   = mw * 32 + (lane & 15);
  const int bnc  = nw * 64 + (lane & 15);
  const int kb   = half * 8;

  for (int k0 = 0; k0 < K; k0 += BK) {
    // ---- stage A tile (fp32 -> bf16 via VGPRs) ----
    {
      int grow = m0 + arow;
      unsigned short tmp[16];
      if (grow < M) {
        const v4f* ap4 = (const v4f*)(A + (size_t)grow * lda + k0 + aseg);
        #pragma unroll
        for (int q = 0; q < 4; ++q) {
          v4f t = ap4[q];
          tmp[q*4+0] = f2bf(t.x); tmp[q*4+1] = f2bf(t.y);
          tmp[q*4+2] = f2bf(t.z); tmp[q*4+3] = f2bf(t.w);
        }
      } else {
        #pragma unroll
        for (int q = 0; q < 16; ++q) tmp[q] = 0;
      }
      #pragma unroll
      for (int q = 0; q < 16; ++q) Asl[arow * LDSROW + aseg + q] = tmp[q];
    }
    // ---- stage B tile: async global->LDS, bypassing VGPRs (ASYNCcnt) ----
    // 128 rows x 64B = 512 x 16B chunks; 2 chunks per thread.
    #pragma unroll
    for (int i = 0; i < 2; ++i) {
      int chunk = tid + i * 256;
      int brow  = chunk >> 2;
      int bseg  = (chunk & 3) * 8;                       // ushort offset in row
      unsigned ldsoff = (unsigned)(uintptr_t)&Bsl[brow * LDSROW + bseg];
      const unsigned short* gp = Bt + (size_t)(n0 + brow) * K + k0 + bseg;
      asm volatile("global_load_async_to_lds_b128 %0, %1, off"
                   :: "v"(ldsoff), "v"(gp) : "memory");
    }
    asm volatile("s_wait_asynccnt 0x0" ::: "memory");
    __syncthreads();

    // ---- A fragments (16x32 bf16, ISA lane layout) ----
    FragU af0, af1;
    af0.u[0] = *(const v4u*)&Asl[fm * LDSROW + half * 8];
    af0.u[1] = *(const v4u*)&Asl[fm * LDSROW + 16 + half * 8];
    af1.u[0] = *(const v4u*)&Asl[(fm + 16) * LDSROW + half * 8];
    af1.u[1] = *(const v4u*)&Asl[(fm + 16) * LDSROW + 16 + half * 8];

    #pragma unroll
    for (int j = 0; j < 4; ++j) {
      FragU bfu;
      const unsigned short* bp = &Bsl[(bnc + j * 16) * LDSROW + kb];
      bfu.u[0] = *(const v4u*)(bp);
      bfu.u[1] = *(const v4u*)(bp + 16);
      acc[0][j] = __builtin_amdgcn_wmma_f32_16x16x32_bf16(
          false, af0.b, false, bfu.b, (short)0, acc[0][j], false, false);
      acc[1][j] = __builtin_amdgcn_wmma_f32_16x16x32_bf16(
          false, af1.b, false, bfu.b, (short)0, acc[1][j], false, false);
    }
    __syncthreads();
  }

  // ---- epilogue: C/D 16x16 f32 layout ----
  const int colbase = n0 + nw * 64 + (lane & 15);
  const int rowwave = m0 + mw * 32 + (lane >> 4) * 8;
  #pragma unroll
  for (int i = 0; i < 2; ++i) {
    #pragma unroll
    for (int j = 0; j < 4; ++j) {
      int col = colbase + j * 16;
      float bv = (flags & 1) ? bias[col] : 0.0f;
      #pragma unroll
      for (int r = 0; r < 8; ++r) {
        int row = rowwave + i * 16 + r;
        if (row < M) {
          float v = acc[i][j][r] + bv;
          if (flags & 2) v = leakyf(v);
          C[(size_t)row * ldc + col] = v;
        }
      }
    }
  }
}

// ---------------- attention: per-edge per-head score + segment max ----------------
__global__ void k_scores(const float* __restrict__ q, const float* __restrict__ kf,
                         const float* __restrict__ ek,
                         const int* __restrict__ src, const int* __restrict__ dst,
                         const int* __restrict__ bt,
                         float* __restrict__ scores, unsigned int* __restrict__ nmax,
                         int nE)
{
  int idx = blockIdx.x * blockDim.x + threadIdx.x;
  if (idx >= nE * NUM_HEAD) return;
  int e = idx >> 3, h = idx & 7;
  int s = src[e], d = dst[e], t = bt[e];
  const float* qp = q  + (size_t)d * NFEAT + h * HEAD_D;
  const float* kp = kf + (size_t)s * NFEAT + h * HEAD_D;
  const float* ep = ek + (size_t)t * NFEAT + h * HEAD_D;
  float acc = 0.f;
  #pragma unroll
  for (int i = 0; i < HEAD_D; ++i) acc += qp[i] * (kp[i] + ep[i]);
  acc *= 0.17677669529663687f;                  // 1/sqrt(32)
  acc = leakyf(acc);
  scores[idx] = acc;
  atomicMax(&nmax[(size_t)d * NUM_HEAD + h], encf(acc));
}

__global__ void k_expsum(float* __restrict__ scores, const unsigned int* __restrict__ nmax,
                         float* __restrict__ denom, const int* __restrict__ dst, int nE)
{
  int idx = blockIdx.x * blockDim.x + threadIdx.x;
  if (idx >= nE * NUM_HEAD) return;
  int e = idx >> 3, h = idx & 7;
  size_t seg = (size_t)dst[e] * NUM_HEAD + h;
  float m  = decf(nmax[seg]);
  float ex = __expf(scores[idx] - m);
  scores[idx] = ex;
  atomicAdd(&denom[seg], ex);
}

__global__ void k_attn_agg(const float* __restrict__ ex, const float* __restrict__ denom,
                           const float* __restrict__ v,
                           const int* __restrict__ src, const int* __restrict__ dst,
                           float* __restrict__ out, int nE)
{
  int idx = blockIdx.x * blockDim.x + threadIdx.x;
  if (idx >= nE * NUM_HEAD) return;
  int e = idx >> 3, h = idx & 7;
  float alpha = ex[idx] / (denom[(size_t)dst[e] * NUM_HEAD + h] + 1e-9f);
  const float* vp = v + (size_t)src[e] * NFEAT + h * HEAD_D;
  float*       op = out + (size_t)dst[e] * NFEAT + h * HEAD_D;
  #pragma unroll
  for (int i = 0; i < HEAD_D; ++i) atomicAdd(&op[i], alpha * vp[i]);
}

// ---------------- RGC per-relation scatter: agg[dst] += hW[src] ----------------
__global__ void k_rgc_scatter(const int* __restrict__ etype, const int* __restrict__ src,
                              const int* __restrict__ dst, const float* __restrict__ hW,
                              float* __restrict__ agg, int dcol, int r)
{
  int e = blockIdx.x;
  if (etype[e] != r) return;
  const float* hp = hW + (size_t)src[e] * dcol;
  float*       ap = agg + (size_t)dst[e] * dcol;
  for (int c = threadIdx.x; c < dcol; c += blockDim.x)
    atomicAdd(&ap[c], hp[c]);
}

// ---------------- LayerNorm (wave32 per row) ----------------
// flags: bit0 pre-leaky, bit1 post-leaky. addvec[N]/addmat[M,N] optional (added after LN affine).
__global__ void k_layernorm(const float* __restrict__ X, int ldx,
                            float* __restrict__ Y, int ldy,
                            const float* __restrict__ g, const float* __restrict__ b,
                            const float* __restrict__ addvec,
                            const float* __restrict__ addmat, int ldam,
                            int M, int N, int flags)
{
  int row = blockIdx.x * (blockDim.x >> 5) + (threadIdx.x >> 5);
  int lane = threadIdx.x & 31;
  if (row >= M) return;
  const float* xr = X + (size_t)row * ldx;
  float s = 0.f, ss = 0.f;
  for (int c = lane; c < N; c += 32) {
    float v = xr[c];
    if (flags & 1) v = leakyf(v);
    s += v; ss += v * v;
  }
  #pragma unroll
  for (int o = 16; o >= 1; o >>= 1) {
    s  += __shfl_xor(s,  o, 32);
    ss += __shfl_xor(ss, o, 32);
  }
  float invN = 1.f / (float)N;
  float mu   = s * invN;
  float var  = ss * invN - mu * mu;
  float inv  = rsqrtf(var + LN_EPS);
  float* yr  = Y + (size_t)row * ldy;
  for (int c = lane; c < N; c += 32) {
    float v = xr[c];
    if (flags & 1) v = leakyf(v);
    float o = (v - mu) * inv * g[c] + b[c];
    if (addvec) o += addvec[c];
    if (addmat) o += addmat[(size_t)row * ldam + c];
    if (flags & 2) o = leakyf(o);
    yr[c] = o;
  }
}

// ---------------- output head: (M,256) x (256,4) + b ----------------
__global__ void k_out(const float* __restrict__ Z, const float* __restrict__ W,
                      const float* __restrict__ b, float* __restrict__ out, int M)
{
  int n = blockIdx.x * blockDim.x + threadIdx.x;
  if (n >= M) return;
  float a0 = b[0], a1 = b[1], a2 = b[2], a3 = b[3];
  const float* zr = Z + (size_t)n * 256;
  for (int kk = 0; kk < 256; ++kk) {
    float z = zr[kk];
    const float* w = W + kk * 4;
    a0 += z * w[0]; a1 += z * w[1]; a2 += z * w[2]; a3 += z * w[3];
  }
  float* o = out + (size_t)n * 4;
  o[0] = a0; o[1] = a1; o[2] = a2; o[3] = a3;
}

// ==================================================================
extern "C" void kernel_launch(void* const* d_in, const int* in_sizes, int n_in,
                              void* d_out, int out_size, void* d_ws, size_t ws_size,
                              hipStream_t stream)
{
  (void)in_sizes; (void)n_in; (void)out_size; (void)ws_size;

  // ---- inputs (flattened setup_inputs() dict order) ----
  const int*   atom_type = (const int*)d_in[0];
  const int*   bond_type = (const int*)d_in[1];
  const int*   src       = (const int*)d_in[2];
  const int*   dst       = (const int*)d_in[3];
  const float* atom_emb  = (const float*)d_in[4];   // (101,256)
  const float* bond_emb  = (const float*)d_in[5];   // (5,256)
  const float* skip_W    = (const float*)d_in[6];   // (1024,256)
  const float* skip_g    = (const float*)d_in[7];
  const float* skip_b    = (const float*)d_in[8];
  // attn layer l: d_in[9+5l .. +4] = Wq,Wk,Wv,We,Wo
  // rgc  layer i: d_in[24+5i .. +4] = W(4,di,do), W_loop, bias, ln_g, ln_b
  // mlp  layer i: d_in[44+4i .. +3] = W, b, ln_g, ln_b
  const float* out_W = (const float*)d_in[60];      // (256,4)
  const float* out_b = (const float*)d_in[61];

  const int M = N_NODES, nE = N_EDGES;

  // ---- workspace bump allocator ----
  char* cur = (char*)d_ws;
  auto alloc = [&](size_t bytes) -> void* {
    void* p = cur; cur += (bytes + 255) & ~(size_t)255; return p;
  };

  float* skipbuf  = (float*)alloc((size_t)M * 1024 * 4);      // [x | h1 | h2 | h3]
  float* zbuf     = (float*)alloc((size_t)M * 1664 * 4);      // [t | r1 | r2 | r3 | r4]
  float* qb       = (float*)alloc((size_t)M * NFEAT * 4);
  float* kb       = (float*)alloc((size_t)M * NFEAT * 4);
  float* vb       = (float*)alloc((size_t)M * NFEAT * 4);
  float* tmp256   = (float*)alloc((size_t)M * NFEAT * 4);     // attn agg / skip tmp
  float* scores   = (float*)alloc((size_t)nE * NUM_HEAD * 4);
  unsigned int* nmax = (unsigned int*)alloc((size_t)M * NUM_HEAD * 4);
  float* denom    = (float*)alloc((size_t)M * NUM_HEAD * 4);
  float* ekbuf    = (float*)alloc((size_t)16 * NFEAT * 4);
  float* agg      = (float*)alloc((size_t)M * 512 * 4);
  float* tmpbuf   = (float*)alloc((size_t)M * 512 * 4);       // hW / mlp gemm out
  float* actbuf   = (float*)alloc((size_t)M * 512 * 4);       // W_loop out / mlp act

  auto wt_alloc = [&](size_t nel) -> unsigned short* {
    return (unsigned short*)alloc(nel * 2);
  };
  auto prep = [&](const float* W, unsigned short* Wt, int K, int N) {
    int total = K * N;
    k_prep_w<<<dim3((total + 255) / 256), dim3(256), 0, stream>>>(W, Wt, K, N);
  };
  auto gemm = [&](const float* A, int lda, const unsigned short* Bt, const float* bias,
                  float* C, int ldc, int Mm, int Nn, int Kk, int flags) {
    dim3 grid(Nn / BN, (Mm + BM - 1) / BM);
    k_gemm<<<grid, dim3(256), 0, stream>>>(A, lda, Bt, bias, C, ldc, Mm, Nn, Kk, flags);
  };

  // ---- transpose+convert all GEMM weights to bf16 (N,K) ----
  unsigned short* attnWt[3][5];
  for (int l = 0; l < 3; ++l)
    for (int w = 0; w < 5; ++w) {
      attnWt[l][w] = wt_alloc((size_t)NFEAT * NFEAT);
      prep((const float*)d_in[9 + 5 * l + w], attnWt[l][w], NFEAT, NFEAT);
    }
  unsigned short* skipWt = wt_alloc((size_t)1024 * 256);
  prep(skip_W, skipWt, 1024, 256);

  const int rgc_di[4] = {256, 512, 384, 256};
  const int rgc_do[4] = {512, 384, 256, 256};
  unsigned short* rgcWt[4];
  unsigned short* rgcLoopWt[4];
  for (int i = 0; i < 4; ++i) {
    size_t slice = (size_t)rgc_di[i] * rgc_do[i];
    rgcWt[i] = wt_alloc(4 * slice);
    const float* Wr = (const float*)d_in[24 + 5 * i + 0];
    for (int r = 0; r < 4; ++r)
      prep(Wr + r * slice, rgcWt[i] + r * slice, rgc_di[i], rgc_do[i]);
    rgcLoopWt[i] = wt_alloc(slice);
    prep((const float*)d_in[24 + 5 * i + 1], rgcLoopWt[i], rgc_di[i], rgc_do[i]);
  }

  const int mlp_di[4] = {1664, 256, 384, 512};
  const int mlp_do[4] = {256, 384, 512, 256};
  unsigned short* mlpWt[4];
  for (int i = 0; i < 4; ++i) {
    mlpWt[i] = wt_alloc((size_t)mlp_di[i] * mlp_do[i]);
    prep((const float*)d_in[44 + 4 * i + 0], mlpWt[i], mlp_di[i], mlp_do[i]);
  }

  // ---- 1. node embedding into skipbuf cols [0,256) ----
  k_embed<<<dim3(((size_t)M * NFEAT + 255) / 256), dim3(256), 0, stream>>>(
      atom_type, atom_emb, skipbuf, 1024, M);

  // ---- 2. attention layers ----
  const int eh = nE * NUM_HEAD;
  for (int l = 0; l < 3; ++l) {
    const float* h = skipbuf + l * NFEAT;              // lda = 1024
    gemm(h, 1024, attnWt[l][0], nullptr, qb, NFEAT, M, NFEAT, NFEAT, 0);
    gemm(h, 1024, attnWt[l][1], nullptr, kb, NFEAT, M, NFEAT, NFEAT, 0);
    gemm(h, 1024, attnWt[l][2], nullptr, vb, NFEAT, M, NFEAT, NFEAT, 0);
    // ek table: only 4 distinct edge-feature rows
    gemm(bond_emb, NFEAT, attnWt[l][3], nullptr, ekbuf, NFEAT, 4, NFEAT, NFEAT, 0);

    hipMemsetAsync(nmax,  0, (size_t)M * NUM_HEAD * 4, stream);
    hipMemsetAsync(denom, 0, (size_t)M * NUM_HEAD * 4, stream);
    hipMemsetAsync(tmp256, 0, (size_t)M * NFEAT * 4, stream);

    dim3 eg((eh + 255) / 256);
    k_scores<<<eg, dim3(256), 0, stream>>>(qb, kb, ekbuf, src, dst, bond_type,
                                           scores, nmax, nE);
    k_expsum<<<eg, dim3(256), 0, stream>>>(scores, nmax, denom, dst, nE);
    k_attn_agg<<<eg, dim3(256), 0, stream>>>(scores, denom, vb, src, dst, tmp256, nE);

    // Wo projection -> next skip slot
    gemm(tmp256, NFEAT, attnWt[l][4], nullptr,
         skipbuf + (l + 1) * NFEAT, 1024, M, NFEAT, NFEAT, 0);
  }

  // ---- 3. skip concat -> linear -> leaky -> LN -> zbuf cols [0,256) ----
  gemm(skipbuf, 1024, skipWt, nullptr, tmp256, NFEAT, M, 256, 1024, 0);
  k_layernorm<<<dim3((M + 7) / 8), dim3(256), 0, stream>>>(
      tmp256, NFEAT, zbuf, 1664, skip_g, skip_b, nullptr, nullptr, 0, M, 256, 1);

  // ---- 4. relational graph conv layers ----
  const int zin[4]  = {0, 256, 768, 1152};
  const int zout[4] = {256, 768, 1152, 1408};
  for (int i = 0; i < 4; ++i) {
    int di = rgc_di[i], dd = rgc_do[i];
    const float* hin = zbuf + zin[i];                  // lda = 1664
    hipMemsetAsync(agg, 0, (size_t)M * dd * 4, stream);
    size_t slice = (size_t)di * dd;
    for (int r = 0; r < 4; ++r) {
      gemm(hin, 1664, rgcWt[i] + r * slice, nullptr, tmpbuf, dd, M, dd, di, 0);
      k_rgc_scatter<<<dim3(nE), dim3(256), 0, stream>>>(
          bond_type, src, dst, tmpbuf, agg, dd, r);
    }
    gemm(hin, 1664, rgcLoopWt[i], nullptr, actbuf, dd, M, dd, di, 0);
    // out = leaky( LN(agg)*g+b + bias + h@W_loop )
    k_layernorm<<<dim3((M + 7) / 8), dim3(256), 0, stream>>>(
        agg, dd, zbuf + zout[i], 1664,
        (const float*)d_in[24 + 5 * i + 3], (const float*)d_in[24 + 5 * i + 4],
        (const float*)d_in[24 + 5 * i + 2], actbuf, dd, M, dd, 2);
  }

  // ---- 5. MLP over concat z (1664) ----
  const float* cin = zbuf; int lda = 1664, Kc = 1664;
  for (int i = 0; i < 4; ++i) {
    int dd = mlp_do[i];
    gemm(cin, lda, mlpWt[i], (const float*)d_in[44 + 4 * i + 1],
         tmpbuf, dd, M, dd, Kc, 1 /*bias*/);
    k_layernorm<<<dim3((M + 7) / 8), dim3(256), 0, stream>>>(
        tmpbuf, dd, actbuf, dd,
        (const float*)d_in[44 + 4 * i + 2], (const float*)d_in[44 + 4 * i + 3],
        nullptr, nullptr, 0, M, dd, 1 /*pre-leaky*/);
    cin = actbuf; lda = dd; Kc = dd;
  }

  // ---- 6. output head ----
  k_out<<<dim3((M + 255) / 256), dim3(256), 0, stream>>>(
      actbuf, out_W, out_b, (float*)d_out, M);
}
